// FeedForwardQuantum_65481071409132
// MI455X (gfx1250) — matrix operand order repaired and checked by reading, not compiled, not run on hardware
//
#include <hip/hip_runtime.h>
#include <hip/hip_bf16.h>
#include <math.h>

// ---------------- problem constants (from reference) ----------------
#define B_   8
#define S_   4096
#define E_   1024
#define NQ_  10
#define F_   1024
#define M_   (B_ * S_)       // 32768 tokens

// ---------------- vector types ----------------
typedef __attribute__((ext_vector_type(16))) __bf16          bf16x16;
typedef __attribute__((ext_vector_type(8)))  float           v8f;
typedef __attribute__((ext_vector_type(8)))  unsigned short  us8;
typedef __attribute__((ext_vector_type(4)))  unsigned short  us4;
typedef __attribute__((ext_vector_type(4)))  float           f32x4;
typedef __attribute__((ext_vector_type(4)))  unsigned int    ui32x4;
typedef __attribute__((ext_vector_type(8)))  int             i32x8;
typedef __attribute__((ext_vector_type(4)))  int             i32x4;

// ---------------- feature detection (device pass only) ----------------
#if defined(__gfx1250__) && __has_builtin(__builtin_amdgcn_tensor_load_to_lds) && \
    __has_builtin(__builtin_amdgcn_s_wait_tensorcnt)
#define HAVE_TDM 1
#else
#define HAVE_TDM 0
#endif

#if __has_include(<hip/amd_detail/amd_gfx1250_TDM.h>)
#define TDM_6ARG 1   // therock-10.0 headers -> 6-arg builtin
#else
#define TDM_6ARG 0   // ROCm 7.2 -> 5-arg builtin
#endif

static __device__ __forceinline__ unsigned short f32_to_bf16_rne(float f) {
    unsigned int u = __float_as_uint(f);
    unsigned int r = (u + 0x7FFFu + ((u >> 16) & 1u)) >> 16;
    return (unsigned short)r;
}

// ---------------- kernel 0: W2 fp32 -> bf16 ----------------
__global__ void w2_to_bf16_kernel(const float* __restrict__ w2,
                                  unsigned short* __restrict__ w2b) {
    int i = (blockIdx.x * 256 + threadIdx.x) * 4;
    f32x4 v = *(const f32x4*)(w2 + i);
    us4 o;
    o[0] = f32_to_bf16_rne(v[0]);
    o[1] = f32_to_bf16_rne(v[1]);
    o[2] = f32_to_bf16_rne(v[2]);
    o[3] = f32_to_bf16_rne(v[3]);
    *(us4*)(w2b + i) = o;
}

// ---------------- kernel 1: z = cos(x[:, :NQ]); h = relu(z @ W1^T + b1) -> bf16 ----------------
__global__ void h_kernel(const float* __restrict__ x,
                         const float* __restrict__ W1,   // [F, NQ] row-major
                         const float* __restrict__ b1,   // [F]
                         unsigned short* __restrict__ h) // [M, F] bf16
{
    __shared__ float z[NQ_];
    const int m = blockIdx.x;
    const int t = threadIdx.x;
    if (t < NQ_) z[t] = cosf(x[(size_t)m * E_ + t]);
    __syncthreads();

    float zr[NQ_];
#pragma unroll
    for (int q = 0; q < NQ_; ++q) zr[q] = z[q];

    const int f0 = t * 4;
    us4 o;
#pragma unroll
    for (int j = 0; j < 4; ++j) {
        const int f = f0 + j;
        float acc = b1[f];
#pragma unroll
        for (int q = 0; q < NQ_; ++q) acc = fmaf(zr[q], W1[f * NQ_ + q], acc);
        acc = fmaxf(acc, 0.0f);
        o[j] = f32_to_bf16_rne(acc);
    }
    *(us4*)(h + (size_t)m * F_ + f0) = o;
}

// ---------------- kernel 2: out = h @ W2^T + b2 via bf16 WMMA ----------------
#define BM 64
#define BN 256
#define BK 32
#define LSTR 40   // ushorts per LDS row (32 data + 8 pad) = 80 bytes, 16B-aligned rows
#define KT (F_ / BK)

#if HAVE_TDM
// Issue one TDM descriptor: 2D tile [tile_dim1 rows x BK cols] of a row-major
// [tensor_dim1 x F_] bf16 tensor, starting at gptr, into LDS at lds_byte_addr.
// pad_enable: +16B after every 64B row -> 80B LDS row stride (matches LSTR).
static __device__ __forceinline__ void tdm_load_tile(const unsigned short* gptr,
                                                     unsigned lds_byte_addr,
                                                     unsigned tensor_dim1,
                                                     unsigned tile_dim1)
{
    const unsigned long long ga = (unsigned long long)(uintptr_t)gptr;
    ui32x4 g0;
    g0[0] = 1u;                                                   // count=1, user mode
    g0[1] = lds_byte_addr;                                        // lds_addr
    g0[2] = (unsigned)ga;                                         // global_addr[31:0]
    g0[3] = (unsigned)((ga >> 32) & 0x01FFFFFFull) | (2u << 30);  // global_addr[56:32] | type=2

    i32x8 g1;
    g1[0] = (int)((1u << 16)      // data_size = 2 bytes
                | (1u << 20)      // pad_enable
                | (3u << 22)      // pad_interval: after 16 DWORDs (64B)
                | (3u << 25));    // pad_amount: 4 DWORDs (16B)
    g1[1] = (int)(((unsigned)F_ & 0xFFFFu) << 16);                  // tensor_dim0[15:0]
    g1[2] = (int)((tensor_dim1 & 0xFFFFu) << 16);                   // dim0[31:16]=0 | dim1[15:0]
    g1[3] = (int)(((tensor_dim1 >> 16) & 0xFFFFu) | ((unsigned)BK << 16)); // dim1[31:16] | tile_dim0
    g1[4] = (int)(tile_dim1 & 0xFFFFu);                             // tile_dim1 | tile_dim2=0
    g1[5] = (int)F_;                                                // tensor_dim0_stride[31:0]
    g1[6] = 0;                                                      // stride0 hi | stride1 lo
    g1[7] = 0;                                                      // stride1 hi

    i32x4 z4 = {};
#if TDM_6ARG
    i32x8 z8 = {};
    __builtin_amdgcn_tensor_load_to_lds(g0, g1, z4, z4, z8, 0);
#else
    __builtin_amdgcn_tensor_load_to_lds(g0, g1, z4, z4, 0);
#endif
}
#endif // HAVE_TDM

// 8x WMMA on one (At, Bt) K-slice; fragment gathers per ISA 7.12.2 16-bit layouts.
static __device__ __forceinline__ void compute_tile(const unsigned short* __restrict__ At,
                                                    const unsigned short* __restrict__ Bt,
                                                    int waveM, int waveN, int hl, int l,
                                                    v8f (&acc)[2][4])
{
    union Frag { us8 u[2]; bf16x16 v; };
    Frag af[2], bf[4];
#pragma unroll
    for (int mi = 0; mi < 2; ++mi) {
        // A 16x32: lane<16 -> row M=l, K 0..7 (V0-3) & 16..23 (V4-7); lane>=16 -> K 8..15 & 24..31
        const unsigned short* ap = At + (waveM * 32 + mi * 16 + l) * LSTR;
        af[mi].u[0] = *(const us8*)(ap + hl * 8);
        af[mi].u[1] = *(const us8*)(ap + 16 + hl * 8);
    }
#pragma unroll
    for (int ni = 0; ni < 4; ++ni) {
        // B 32x16: lane<16 -> col N=l, K 0..7 & 16..23; lane>=16 -> col N=l, K 8..15 & 24..31
        const unsigned short* bp = Bt + (waveN * 64 + ni * 16 + l) * LSTR;
        bf[ni].u[0] = *(const us8*)(bp + hl * 8);
        bf[ni].u[1] = *(const us8*)(bp + 16 + hl * 8);
    }
#pragma unroll
    for (int mi = 0; mi < 2; ++mi)
#pragma unroll
        for (int ni = 0; ni < 4; ++ni)
            acc[mi][ni] = __builtin_amdgcn_wmma_f32_16x16x32_bf16(
                false, af[mi].v, false, bf[ni].v,
                (short)0, acc[mi][ni], false, false);
}

__global__ __launch_bounds__(256) void gemm2_kernel(
        const unsigned short* __restrict__ h,    // [M, F] bf16
        const unsigned short* __restrict__ w2b,  // [E, F] bf16
        const float* __restrict__ b2,            // [E]
        float* __restrict__ out)                 // [M, E] fp32
{
    const int tid   = threadIdx.x;
    const int wave  = tid >> 5;       // 0..7
    const int lane  = tid & 31;
    const int hl    = lane >> 4;      // half-wave: 0 or 1
    const int l     = lane & 15;
    const int waveM = wave & 1;       // 2 waves in M  -> 32 rows each
    const int waveN = wave >> 1;      // 4 waves in N  -> 64 cols each
    const int m0    = blockIdx.y * BM;
    const int n0    = blockIdx.x * BN;

    v8f acc[2][4] = {};

#if HAVE_TDM
    // ---- double-buffered LDS, filled by the Tensor Data Mover ----
    __shared__ __align__(16) unsigned short At[2][BM * LSTR];  // 2 x  5.0 KB
    __shared__ __align__(16) unsigned short Bt[2][BN * LSTR];  // 2 x 20.0 KB

    const unsigned atl[2] = { (unsigned)(uintptr_t)&At[0][0], (unsigned)(uintptr_t)&At[1][0] };
    const unsigned btl[2] = { (unsigned)(uintptr_t)&Bt[0][0], (unsigned)(uintptr_t)&Bt[1][0] };

    if (wave == 0) {   // one wave issues DMA; TDM ignores EXEC, 1 op per instruction
        tdm_load_tile(h   + (size_t)m0 * F_, atl[0], (unsigned)M_, BM);
        tdm_load_tile(w2b + (size_t)n0 * F_, btl[0], (unsigned)E_, BN);
    }

    for (int kt = 0; kt < KT; ++kt) {
        const int cur = kt & 1;
        const int nxt = cur ^ 1;
        if (wave == 0) {
            if (kt + 1 < KT) {
                const int k2 = (kt + 1) * BK;
                tdm_load_tile(h   + (size_t)m0 * F_ + k2, atl[nxt], (unsigned)M_, BM);
                tdm_load_tile(w2b + (size_t)n0 * F_ + k2, btl[nxt], (unsigned)E_, BN);
                __builtin_amdgcn_s_wait_tensorcnt(2);   // tile kt complete, kt+1 in flight
            } else {
                __builtin_amdgcn_s_wait_tensorcnt(0);
            }
        }
        __syncthreads();                                // publish buf[cur] to all waves
        compute_tile(At[cur], Bt[cur], waveM, waveN, hl, l, acc);
        __syncthreads();                                // everyone done reading buf[cur]
    }
#else
    // ---- fallback: single-buffer manual global->LDS path ----
    __shared__ __align__(16) unsigned short At[BM * LSTR];
    __shared__ __align__(16) unsigned short Bt[BN * LSTR];

    const int arow = tid >> 2;          // 0..63
    const int acol = (tid & 3) * 8;     // 0,8,16,24
    const unsigned short* ag = h   + (size_t)(m0 + arow) * F_ + acol;
    const unsigned short* bg = w2b + (size_t)(n0 + tid)  * F_;

    for (int k = 0; k < F_; k += BK) {
        us8 av  = *(const us8*)(ag + k);
        us8 bv0 = *(const us8*)(bg + k);
        us8 bv1 = *(const us8*)(bg + k + 8);
        us8 bv2 = *(const us8*)(bg + k + 16);
        us8 bv3 = *(const us8*)(bg + k + 24);

        __syncthreads();
        *(us8*)(At + arow * LSTR + acol) = av;
        {
            unsigned short* bp = Bt + tid * LSTR;
            *(us8*)(bp +  0) = bv0;
            *(us8*)(bp +  8) = bv1;
            *(us8*)(bp + 16) = bv2;
            *(us8*)(bp + 24) = bv3;
        }
        __syncthreads();
        compute_tile(At, Bt, waveM, waveN, hl, l, acc);
    }
#endif

    // ---- epilogue: C/D layout -> element (VGPR p, lane) = C[M = p + 8*hl][N = l] ----
#pragma unroll
    for (int mi = 0; mi < 2; ++mi) {
#pragma unroll
        for (int ni = 0; ni < 4; ++ni) {
            const int e    = n0 + waveN * 64 + ni * 16 + l;
            const float bv = b2[e];
#pragma unroll
            for (int p = 0; p < 8; ++p) {
                const int m = m0 + waveM * 32 + mi * 16 + hl * 8 + p;
                out[(size_t)m * E_ + e] = acc[mi][ni][p] + bv;
            }
        }
    }
}

// ---------------- host launcher ----------------
extern "C" void kernel_launch(void* const* d_in, const int* in_sizes, int n_in,
                              void* d_out, int out_size, void* d_ws, size_t ws_size,
                              hipStream_t stream) {
    const float* x  = (const float*)d_in[0];
    // d_in[1] = rz_theta: mathematically drops out of <Z>; unused.
    const float* W1 = (const float*)d_in[2];
    const float* b1 = (const float*)d_in[3];
    const float* W2 = (const float*)d_in[4];
    const float* b2 = (const float*)d_in[5];
    float* out = (float*)d_out;

    // workspace layout: [0, 2MB) W2 bf16 ; [2MB, 66MB) h bf16
    unsigned short* w2b  = (unsigned short*)d_ws;
    unsigned short* hbuf = (unsigned short*)((char*)d_ws +
                           (size_t)E_ * F_ * sizeof(unsigned short));

    w2_to_bf16_kernel<<<(E_ * F_) / 1024, 256, 0, stream>>>(W2, w2b);
    h_kernel<<<M_, 256, 0, stream>>>(x, W1, b1, hbuf);

    dim3 grid(E_ / BN, M_ / BM);
    gemm2_kernel<<<grid, 256, 0, stream>>>(hbuf, w2b, b2, out);
}
